// HierarchicalDynamicGAT_67422396613192
// MI455X (gfx1250) — compile-verified, matrix operand order checked by reading, not run on hardware
//
#include <hip/hip_runtime.h>
#include <hip/hip_bf16.h>
#include <math.h>

#define N_NODES 50000
#define N_EDGES 800000
#define E_TOT   (N_EDGES + N_NODES)   // reference adds self loops
#define IN_DIM  128
#define HID     64
#define HEADS   4
#define C1      (HEADS * HID)         // 256
#define OUT_DIM 64
#define NEG_SLOPE 0.2f

typedef __attribute__((ext_vector_type(16))) __bf16 v16bf;
typedef __attribute__((ext_vector_type(8)))  __bf16 v8bf;
typedef __attribute__((ext_vector_type(8)))  float  v8f;

// ---------- helpers ----------
static __device__ __forceinline__ int enc_f32(float f) {
    int i = __float_as_int(f);
    return i >= 0 ? i : (i ^ 0x7fffffff);   // order-preserving float->int
}
static __device__ __forceinline__ float dec_f32(int i) {
    return __int_as_float(i >= 0 ? i : (i ^ 0x7fffffff));
}
static __device__ __forceinline__ void edge_sd(const int* __restrict__ ei, int e, int& s, int& d) {
    if (e < N_EDGES) { s = ei[e]; d = ei[N_EDGES + e]; }
    else             { s = d = e - N_EDGES; }          // self loop
}

// ---------- casts / transposes ----------
__global__ void k_cast_bf16(const float* __restrict__ src, __bf16* __restrict__ dst, size_t n) {
    size_t i = (size_t)blockIdx.x * blockDim.x + threadIdx.x;
    if (i < n) dst[i] = (__bf16)src[i];
}
// W is K x N row-major; Wt is N x K row-major (bf16)
__global__ void k_transpose_cast(const float* __restrict__ W, __bf16* __restrict__ Wt, int K, int N) {
    int i = blockIdx.x * blockDim.x + threadIdx.x;
    if (i < K * N) {
        int k = i / N, n = i % N;
        Wt[(size_t)n * K + k] = (__bf16)W[i];
    }
}

// ---------- WMMA GEMM: C[M,N] = A[M,K] * Bt[N,K]^T  (bf16 in, f32 out) ----------
// One wave per 16x16 tile. A-operand lane layout per CDNA5 ISA 7.12.2:
// lanes 0-15: row m=lane, K in {kb..kb+7, kb+16..kb+23} with kb=k0; lanes 16-31: kb=k0+8.
// B assumed symmetric (per-column striping) -> load from transposed B with identical pattern.
template <int K>
__global__ void k_wmma_gemm(const __bf16* __restrict__ A, const __bf16* __restrict__ Bt,
                            float* __restrict__ C, int Mtiles, int Ntiles) {
    int wave = (blockIdx.x * blockDim.x + threadIdx.x) >> 5;
    int lane = threadIdx.x & 31;
    int tm = wave / Ntiles, tn = wave % Ntiles;
    if (tm >= Mtiles) return;
    int half = lane >> 4;        // K-half selector
    int r    = lane & 15;        // A row / B column within tile
    const __bf16* arow = A  + (size_t)(tm * 16 + r) * K;
    const __bf16* brow = Bt + (size_t)(tn * 16 + r) * K;
    v8f acc = {};
#pragma unroll
    for (int k0 = 0; k0 < K; k0 += 32) {
        int kb = k0 + half * 8;
        if (k0 + 32 < K) {
            __builtin_prefetch(arow + kb + 32, 0, 1);
            __builtin_prefetch(brow + kb + 32, 0, 1);
        }
        v8bf alo = *(const v8bf*)(arow + kb);
        v8bf ahi = *(const v8bf*)(arow + kb + 16);
        v8bf blo = *(const v8bf*)(brow + kb);
        v8bf bhi = *(const v8bf*)(brow + kb + 16);
        v16bf a, b;
#pragma unroll
        for (int i = 0; i < 8; ++i) {
            a[i] = alo[i]; a[8 + i] = ahi[i];
            b[i] = blo[i]; b[8 + i] = bhi[i];
        }
        acc = __builtin_amdgcn_wmma_f32_16x16x32_bf16(
            /*neg_a=*/false, a, /*neg_b=*/false, b,
            /*c_mod=*/(short)0, acc, /*reuse_a=*/false, /*reuse_b=*/false);
    }
    // C/D layout: VGPR v -> M = half*8 + v, N = r
    int N = Ntiles * 16;
    float* cbase = C + (size_t)(tm * 16 + half * 8) * N + tn * 16 + r;
#pragma unroll
    for (int v = 0; v < 8; ++v) cbase[(size_t)v * N] = acc[v];
}

// ---------- attention coefficients: as/ad[n,h] = <H[n,h,:], a_src/dst[h,:]> ----------
__global__ void k_alpha(const float* __restrict__ H, const float* __restrict__ a_src,
                        const float* __restrict__ a_dst, float* __restrict__ as,
                        float* __restrict__ ad, int heads, int C) {
    int i = blockIdx.x * blockDim.x + threadIdx.x;
    if (i >= N_NODES * heads) return;
    int n = i / heads, h = i % heads;
    const float* row = H + ((size_t)n * heads + h) * C;
    float s = 0.f, d = 0.f;
    for (int c = 0; c < C; ++c) {
        float v = row[c];
        s += v * a_src[h * C + c];
        d += v * a_dst[h * C + c];
    }
    as[i] = s; ad[i] = d;
}

// ---------- fills ----------
__global__ void k_fill_i32(int* p, int v, size_t n) {
    size_t i = (size_t)blockIdx.x * blockDim.x + threadIdx.x;
    if (i < n) p[i] = v;
}
__global__ void k_fill_f32(float* p, float v, size_t n) {
    size_t i = (size_t)blockIdx.x * blockDim.x + threadIdx.x;
    if (i < n) p[i] = v;
}

// ---------- segment max over incoming edges (ordered-int atomicMax) ----------
__global__ void k_edge_max(const int* __restrict__ ei, const float* __restrict__ as,
                           const float* __restrict__ ad, int* __restrict__ m, int heads) {
    int i = blockIdx.x * blockDim.x + threadIdx.x;
    if (i >= E_TOT * heads) return;
    int e = i / heads, h = i % heads;
    int s, d; edge_sd(ei, e, s, d);
    float v = as[s * heads + h] + ad[d * heads + h];
    v = v > 0.f ? v : NEG_SLOPE * v;                  // leaky relu
    atomicMax(m + d * heads + h, enc_f32(v));
}

// ---------- exp(e - max) + denominator ----------
__global__ void k_edge_expsum(const int* __restrict__ ei, const float* __restrict__ as,
                              const float* __restrict__ ad, const int* __restrict__ m,
                              float* __restrict__ ex, float* __restrict__ den, int heads) {
    int i = blockIdx.x * blockDim.x + threadIdx.x;
    if (i >= E_TOT * heads) return;
    int e = i / heads, h = i % heads;
    int s, d; edge_sd(ei, e, s, d);
    float v = as[s * heads + h] + ad[d * heads + h];
    v = v > 0.f ? v : NEG_SLOPE * v;
    float x = expf(v - dec_f32(m[d * heads + h]));
    ex[(size_t)e * heads + h] = x;
    atomicAdd(den + d * heads + h, x);
}

// ---------- scatter: agg[dst] += alpha * H[src]  (float4 gathers, f32 atomics) ----------
template <int HEADS_T, int CPH>
__global__ void k_scatter(const int* __restrict__ ei, const float* __restrict__ ex,
                          const float* __restrict__ den, const float* __restrict__ H,
                          float* __restrict__ agg) {
    const int CH  = HEADS_T * CPH;
    const int TPE = CH / 4;                           // threads per edge
    long long tid = (long long)blockIdx.x * blockDim.x + threadIdx.x;
    int e = (int)(tid / TPE);
    if (e >= E_TOT) return;
    int c4 = (int)(tid % TPE) * 4;
    int h  = c4 / CPH;
    int s, d; edge_sd(ei, e, s, d);
    float alpha = ex[(size_t)e * HEADS_T + h] / (den[d * HEADS_T + h] + 1e-16f);
    float4 hv = *(const float4*)(H + (size_t)s * CH + c4);
    float* o = agg + (size_t)d * CH + c4;
    atomicAdd(o + 0, alpha * hv.x);
    atomicAdd(o + 1, alpha * hv.y);
    atomicAdd(o + 2, alpha * hv.z);
    atomicAdd(o + 3, alpha * hv.w);
}

// ---------- bias + ELU + cast to bf16 (feeds GEMM2) ----------
__global__ void k_elu_cast(const float* __restrict__ agg, const float* __restrict__ b,
                           __bf16* __restrict__ out, size_t n, int C) {
    size_t i = (size_t)blockIdx.x * blockDim.x + threadIdx.x;
    if (i >= n) return;
    float v = agg[i] + b[i % C];
    v = v > 0.f ? v : expm1f(v);
    out[i] = (__bf16)v;
}

// ---------- final: out[n] = <agg2[n]+b2, fc_w> + fc_b ----------
__global__ void k_final(const float* __restrict__ agg2, const float* __restrict__ b2,
                        const float* __restrict__ fc_w, const float* __restrict__ fc_b,
                        float* __restrict__ out) {
    int n = blockIdx.x * blockDim.x + threadIdx.x;
    if (n >= N_NODES) return;
    const float* row = agg2 + (size_t)n * OUT_DIM;
    float s = 0.f;
#pragma unroll
    for (int c = 0; c < OUT_DIM; ++c) s += (row[c] + b2[c]) * fc_w[c];
    out[n] = s + fc_b[0];
}

static inline unsigned blk(size_t n, int b) { return (unsigned)((n + (size_t)b - 1) / b); }

extern "C" void kernel_launch(void* const* d_in, const int* in_sizes, int n_in,
                              void* d_out, int out_size, void* d_ws, size_t ws_size,
                              hipStream_t stream) {
    const float* x      = (const float*)d_in[0];
    const int*   ei     = (const int*)  d_in[1];
    const float* W1     = (const float*)d_in[2];
    const float* a_src1 = (const float*)d_in[3];
    const float* a_dst1 = (const float*)d_in[4];
    const float* b1     = (const float*)d_in[5];
    const float* W2     = (const float*)d_in[6];
    const float* a_src2 = (const float*)d_in[7];
    const float* a_dst2 = (const float*)d_in[8];
    const float* b2     = (const float*)d_in[9];
    const float* fc_w   = (const float*)d_in[10];
    const float* fc_b   = (const float*)d_in[11];
    float* out = (float*)d_out;

    // ---- workspace carve-out (256B aligned bump allocator) ----
    char* base = (char*)d_ws;
    size_t cur = 0;
    auto alloc = [&](size_t bytes) -> void* {
        cur = (cur + 255) & ~(size_t)255;
        void* p = base + cur;
        cur += bytes;
        return p;
    };
    __bf16* xb   = (__bf16*)alloc((size_t)N_NODES * IN_DIM * 2);
    __bf16* w1t  = (__bf16*)alloc((size_t)C1 * IN_DIM * 2);
    float*  h1   = (float*) alloc((size_t)N_NODES * C1 * 4);       // dead after scatter1
    float*  as1  = (float*) alloc((size_t)N_NODES * HEADS * 4);
    float*  ad1  = (float*) alloc((size_t)N_NODES * HEADS * 4);
    int*    m1   = (int*)   alloc((size_t)N_NODES * HEADS * 4);
    float*  den1 = (float*) alloc((size_t)N_NODES * HEADS * 4);
    float*  ex1  = (float*) alloc((size_t)E_TOT * HEADS * 4);
    float*  agg1 = (float*) alloc((size_t)N_NODES * C1 * 4);
    __bf16* h1b  = (__bf16*)alloc((size_t)N_NODES * C1 * 2);
    __bf16* w2t  = (__bf16*)alloc((size_t)OUT_DIM * C1 * 2);
    // phase-2 buffers alias the h1 region (29.8MB needed < 51.2MB available)
    float* h2   = h1;
    float* agg2 = h2 + (size_t)N_NODES * OUT_DIM;
    float* ex2  = agg2 + (size_t)N_NODES * OUT_DIM;
    float* as2  = ex2 + E_TOT;
    float* ad2  = as2 + N_NODES;
    int*   m2   = (int*)(ad2 + N_NODES);
    float* den2 = (float*)(m2 + N_NODES);

    const int NEG_INF_ENC = 0x807FFFFF;  // enc_f32(-inf)

    // ================= Layer 1 =================
    k_cast_bf16<<<blk((size_t)N_NODES * IN_DIM, 256), 256, 0, stream>>>(x, xb, (size_t)N_NODES * IN_DIM);
    k_transpose_cast<<<blk((size_t)IN_DIM * C1, 256), 256, 0, stream>>>(W1, w1t, IN_DIM, C1);

    {   // GEMM1: 3125 x 16 tiles, one wave each, 4 waves/block
        int Mtiles = N_NODES / 16, Ntiles = C1 / 16;
        unsigned nblk = blk((size_t)Mtiles * Ntiles * 32, 128);
        k_wmma_gemm<IN_DIM><<<nblk, 128, 0, stream>>>(xb, w1t, h1, Mtiles, Ntiles);
    }
    k_alpha<<<blk((size_t)N_NODES * HEADS, 256), 256, 0, stream>>>(h1, a_src1, a_dst1, as1, ad1, HEADS, HID);

    k_fill_i32<<<blk((size_t)N_NODES * HEADS, 256), 256, 0, stream>>>(m1, NEG_INF_ENC, (size_t)N_NODES * HEADS);
    k_fill_f32<<<blk((size_t)N_NODES * HEADS, 256), 256, 0, stream>>>(den1, 0.f, (size_t)N_NODES * HEADS);
    k_fill_f32<<<blk((size_t)N_NODES * C1, 256), 256, 0, stream>>>(agg1, 0.f, (size_t)N_NODES * C1);

    k_edge_max<<<blk((size_t)E_TOT * HEADS, 256), 256, 0, stream>>>(ei, as1, ad1, m1, HEADS);
    k_edge_expsum<<<blk((size_t)E_TOT * HEADS, 256), 256, 0, stream>>>(ei, as1, ad1, m1, ex1, den1, HEADS);
    k_scatter<HEADS, HID><<<blk((size_t)E_TOT * (C1 / 4), 256), 256, 0, stream>>>(ei, ex1, den1, h1, agg1);

    k_elu_cast<<<blk((size_t)N_NODES * C1, 256), 256, 0, stream>>>(agg1, b1, h1b, (size_t)N_NODES * C1, C1);

    // ================= Layer 2 =================
    k_transpose_cast<<<blk((size_t)C1 * OUT_DIM, 256), 256, 0, stream>>>(W2, w2t, C1, OUT_DIM);
    {   // GEMM2: 3125 x 4 tiles
        int Mtiles = N_NODES / 16, Ntiles = OUT_DIM / 16;
        unsigned nblk = blk((size_t)Mtiles * Ntiles * 32, 128);
        k_wmma_gemm<C1><<<nblk, 128, 0, stream>>>(h1b, w2t, h2, Mtiles, Ntiles);
    }
    k_alpha<<<blk((size_t)N_NODES, 256), 256, 0, stream>>>(h2, a_src2, a_dst2, as2, ad2, 1, OUT_DIM);

    k_fill_i32<<<blk((size_t)N_NODES, 256), 256, 0, stream>>>(m2, NEG_INF_ENC, (size_t)N_NODES);
    k_fill_f32<<<blk((size_t)N_NODES, 256), 256, 0, stream>>>(den2, 0.f, (size_t)N_NODES);
    k_fill_f32<<<blk((size_t)N_NODES * OUT_DIM, 256), 256, 0, stream>>>(agg2, 0.f, (size_t)N_NODES * OUT_DIM);

    k_edge_max<<<blk((size_t)E_TOT, 256), 256, 0, stream>>>(ei, as2, ad2, m2, 1);
    k_edge_expsum<<<blk((size_t)E_TOT, 256), 256, 0, stream>>>(ei, as2, ad2, m2, ex2, den2, 1);
    k_scatter<1, OUT_DIM><<<blk((size_t)E_TOT * (OUT_DIM / 4), 256), 256, 0, stream>>>(ei, ex2, den2, h2, agg2);

    k_final<<<blk((size_t)N_NODES, 256), 256, 0, stream>>>(agg2, b2, fc_w, fc_b, out);
    (void)in_sizes; (void)n_in; (void)out_size; (void)ws_size;
}
